// Block_67826123538913
// MI455X (gfx1250) — matrix operand order repaired and checked by reading, not compiled, not run on hardware
//
#include <hip/hip_runtime.h>

// ============================================================================
// MI455X (gfx1250) implementation of the two-stage attention network.
// GEMMs use v_wmma_f32_16x16x32_f16 (wave32 WMMA). A-tiles are staged into
// double-buffered LDS with CDNA5 async copies (global_load_async_to_lds_b128
// / ASYNCcnt) in a fragment-swizzled layout so every WMMA operand is a
// contiguous 32B load; next tile's fills overlap current tile's WMMAs.
// fp32 accumulation everywhere; f16 only on GEMM operands.
// Requires ws_size >= ~700 MB.
// ============================================================================

typedef __attribute__((ext_vector_type(16))) _Float16 v16h;
typedef __attribute__((ext_vector_type(8)))  float    v8f;

#define BCONST 4
#define TCONST 1024
#define MD 320
#define FF 1280
#define NHD 16
#define HCONST 10
#define HD 32
#define RPEH 64
#define UD (MD*NHD)      // 5120
#define GROWS (BCONST*TCONST)     // 4096
#define LROWS (GROWS*NHD)         // 65536

static __device__ __forceinline__ float lrelu(float v){ return v >= 0.f ? v : 0.2f*v; }

// CDNA5 async global->LDS copy (tracked by ASYNCcnt), per-lane 16 bytes.
static __device__ __forceinline__ void async_load_b128(unsigned lds_off, const _Float16* g){
    asm volatile("global_load_async_to_lds_b128 %0, %1, off"
                 :: "v"(lds_off), "v"(g) : "memory");
}
static __device__ __forceinline__ void wait_asynccnt0(){
    asm volatile("s_wait_asynccnt 0" ::: "memory");
}

// ---------------------------------------------------------------------------
// f32 -> f16 cast
// ---------------------------------------------------------------------------
__global__ void cast_f16_kernel(const float* __restrict__ X, _Float16* __restrict__ Y, size_t n){
    size_t i = (size_t)blockIdx.x*256 + threadIdx.x;
    if (i < n) Y[i] = (_Float16)X[i];
}

// ---------------------------------------------------------------------------
// Repack l_w1 [320,1280] f32 into WMMA B-fragment tile order (f16):
// tile (nt,kt) -> 32 lanes x 16 halves contiguous (32B per lane fragment).
// element (k,n): lane = (n&15) + 16*((k>>3)&1), slot = (k&7) + 8*((k>>4)&1)
// ---------------------------------------------------------------------------
__global__ void repack_w1_kernel(const float* __restrict__ W, _Float16* __restrict__ Wsw){
    int i = blockIdx.x*256 + threadIdx.x;
    if (i >= MD*FF) return;
    int k = i / FF, n = i % FF;
    int kt = k >> 5, kk = k & 31, nt = n >> 4, c = n & 15;
    int hi = (kk >> 3) & 1, j = (kk & 7) + 8*(kk >> 4);
    Wsw[(((size_t)nt*10 + kt)*32 + (c + 16*hi))*16 + j] = (_Float16)W[i];
}

// ---------------------------------------------------------------------------
// Repack activations [Mrows,320] f32 into WMMA A-fragment tile order (f16).
// element (m,k): lane = (m&15) + 16*((k>>3)&1),
//                e = 2*(4*((k>>4)&1) + ((k&6)>>1)) + (k&1)
// ---------------------------------------------------------------------------
__global__ void repack_a_kernel(const float* __restrict__ X, _Float16* __restrict__ Asw,
                                size_t nelem){
    size_t i = (size_t)blockIdx.x*256 + threadIdx.x;
    if (i >= nelem) return;
    int k = (int)(i % MD); int m = (int)(i / MD);
    int mt = m >> 4, rm = m & 15, kt = k >> 5, kk = k & 31;
    int hi = (kk >> 3) & 1;
    int e  = 2*(4*((kk >> 4) & 1) + ((kk & 6) >> 1)) + (kk & 1);
    Asw[(((size_t)mt*10 + kt)*32 + (rm + 16*hi))*16 + e] = (_Float16)X[i];
}

// ---------------------------------------------------------------------------
// LayerNorm over last dim D (one block per row). Optional f32 / f16 outputs.
// ---------------------------------------------------------------------------
__global__ __launch_bounds__(256)
void ln_kernel(const float* __restrict__ X, const float* __restrict__ G,
               const float* __restrict__ Bv, float* __restrict__ Yf,
               _Float16* __restrict__ Yh, int D){
    __shared__ float red[256];
    int row = blockIdx.x, tid = threadIdx.x;
    const float* xr = X + (size_t)row*D;
    float s = 0.f, ss = 0.f;
    for (int i = tid; i < D; i += 256){ float v = xr[i]; s += v; ss += v*v; }
    red[tid] = s; __syncthreads();
    for (int off = 128; off; off >>= 1){ if (tid < off) red[tid] += red[tid+off]; __syncthreads(); }
    float mean = red[0] / D; __syncthreads();
    red[tid] = ss; __syncthreads();
    for (int off = 128; off; off >>= 1){ if (tid < off) red[tid] += red[tid+off]; __syncthreads(); }
    float var = red[0] / D - mean*mean;
    float inv = rsqrtf(var + 1e-5f);
    for (int i = tid; i < D; i += 256){
        float v = (xr[i] - mean) * inv * G[i] + Bv[i];
        if (Yf) Yf[(size_t)row*D + i] = v;
        if (Yh) Yh[(size_t)row*D + i] = (_Float16)v;
    }
}

// ---------------------------------------------------------------------------
// Tiled WMMA GEMM:  C[M,N] = A[M,K] * B[K,N]   (A,B f16 row-major, C f32/f16)
// 256 threads = 8 waves, macro tile 128x64, each wave does 2x2 16x16 tiles.
// Double-buffered LDS: tile t+1 staged (A: async b128 copies into swizzled
// layout, B: scalar swizzle) while tile t feeds the WMMAs.
// Epilogue: optional leaky-relu, optional fp32 residual add (post-lrelu).
// M%128==0, N%64==0, K%32==0 (all shapes here satisfy this).
// ---------------------------------------------------------------------------
template<bool LRELU, bool OUTH>
__global__ __launch_bounds__(256)
void gemm_wmma_kernel(const _Float16* __restrict__ A, const _Float16* __restrict__ Bw,
                      float* __restrict__ Cf, _Float16* __restrict__ Ch,
                      const float* __restrict__ Res, int M, int N, int K){
    // [buf][tile][lane][16 uints(64B); first 8 used] -> per-lane fragment contiguous
    __shared__ __align__(64) unsigned Asw[2][8][32][16];   // 32 KB
    __shared__ __align__(64) unsigned Bsw[2][4][32][16];   // 16 KB
    int tid  = threadIdx.x;
    int lane = tid & 31, wave = tid >> 5;
    int wm = wave & 3, wn = wave >> 2;       // 4 M-waves x 2 N-waves
    int r16 = lane & 15, hi = lane >> 4;
    int bm = blockIdx.x * 128, bn = blockIdx.y * 64;
    unsigned a_base = (unsigned)(uintptr_t)&Asw[0][0][0][0];

    auto stageA = [&](int k0, int buf){
        for (int cc = tid; cc < 512; cc += 256){
            int r = cc >> 2, q = cc & 3;              // row in tile, 16B quad in row
            const _Float16* g = A + (size_t)(bm + r)*K + k0 + q*8;
            unsigned lds = a_base + (unsigned)buf*16384u +
                (unsigned)((((r >> 4)*32 + ((r & 15) + 16*(q & 1)))*64) + (q >> 1)*16);
            async_load_b128(lds, g);
        }
    };
    auto stageB = [&](int k0, int buf){
        _Float16* bs = (_Float16*)&Bsw[buf][0][0][0];
        for (int i = tid; i < 1024; i += 256){
            int k = i >> 5, np = i & 31;
            int n = np << 1;
            const _Float16* src = Bw + (size_t)(k0 + k)*N + bn + n;
            _Float16 v0 = src[0], v1 = src[1];
            int hi2 = (k >> 3) & 1, j = (k & 7) + 8*(k >> 4);
            bs[(((n >> 4)*32 + ((n & 15) + 16*hi2))*32) + j] = v0;
            int n1 = n + 1;
            bs[(((n1 >> 4)*32 + ((n1 & 15) + 16*hi2))*32) + j] = v1;
        }
    };

    v8f acc[2][2] = {};
    int nk = K >> 5;
    stageA(0, 0); stageB(0, 0);
    wait_asynccnt0();
    __syncthreads();
    for (int t = 0; t < nk; ++t){
        int cur = t & 1;
        if (t + 1 < nk){ stageA((t + 1) << 5, cur ^ 1); stageB((t + 1) << 5, cur ^ 1); }
        v16h afr[2], bfr[2];
        #pragma unroll
        for (int ti = 0; ti < 2; ++ti) afr[ti] = *(const v16h*)&Asw[cur][wm*2 + ti][lane][0];
        #pragma unroll
        for (int tj = 0; tj < 2; ++tj) bfr[tj] = *(const v16h*)&Bsw[cur][wn*2 + tj][lane][0];
        #pragma unroll
        for (int ti = 0; ti < 2; ++ti)
            #pragma unroll
            for (int tj = 0; tj < 2; ++tj)
                acc[ti][tj] = __builtin_amdgcn_wmma_f32_16x16x32_f16(
                    false, afr[ti], false, bfr[tj], (short)0, acc[ti][tj], false, false);
        wait_asynccnt0();
        __syncthreads();
    }
    #pragma unroll
    for (int ti = 0; ti < 2; ++ti)
        #pragma unroll
        for (int tj = 0; tj < 2; ++tj)
            #pragma unroll
            for (int r = 0; r < 8; ++r){
                int row = bm + wm*32 + ti*16 + r + 8*hi;
                int col = bn + wn*32 + tj*16 + r16;
                float v = acc[ti][tj][r];
                if (LRELU) v = lrelu(v);
                if (Res)   v += Res[(size_t)row*N + col];
                if (OUTH)  Ch[(size_t)row*N + col] = (_Float16)v;
                else       Cf[(size_t)row*N + col] = v;
            }
}

// ---------------------------------------------------------------------------
// per-head cosine normalization: x *= rsqrt(sum_{32}(x^2) + 1e-6)
// one block (320 thr = 10 waves = 10 heads) per row
// ---------------------------------------------------------------------------
__global__ __launch_bounds__(320)
void qknorm_kernel(float* __restrict__ Q){
    int row = blockIdx.x, tid = threadIdx.x;
    float v = Q[(size_t)row*MD + tid];
    float ss = v*v;
    #pragma unroll
    for (int off = 16; off; off >>= 1) ss += __shfl_xor(ss, off, 32);
    Q[(size_t)row*MD + tid] = v * rsqrtf(ss + 1e-6f);
}

// ---------------------------------------------------------------------------
// Global attention, fused RPE-bias MLP. One block (320 thr) per (b,q) row.
// ---------------------------------------------------------------------------
__global__ __launch_bounds__(320)
void gattn_kernel(const float* __restrict__ Qn, const float* __restrict__ Kn,
                  const float* __restrict__ V,  const float* __restrict__ rel,
                  const float* __restrict__ w1, const float* __restrict__ b1,
                  const float* __restrict__ w2, const float* __restrict__ b2,
                  const float* __restrict__ ls, float* __restrict__ Out){
    __shared__ float sc[HCONST][TCONST];
    __shared__ float qrow[MD];
    __shared__ float w1s[2*RPEH], b1s[RPEH], w2s[RPEH*HCONST], b2s[HCONST], scl[HCONST];
    int row = blockIdx.x, tid = threadIdx.x;
    for (int i = tid; i < 2*RPEH; i += 320) w1s[i] = w1[i];
    for (int i = tid; i < RPEH; i += 320) b1s[i] = b1[i];
    for (int i = tid; i < RPEH*HCONST; i += 320) w2s[i] = w2[i];
    if (tid < HCONST){ b2s[tid] = b2[tid]; scl[tid] = fminf(__expf(ls[tid]), 100.f); }
    qrow[tid] = Qn[(size_t)row*MD + tid];
    __syncthreads();
    // phase 1: RPE bias MLP into sc
    const float* rc = rel + (size_t)row*TCONST*2;
    for (int k = tid; k < TCONST; k += 320){
        float c0 = rc[2*k], c1 = rc[2*k+1];
        float oh[HCONST];
        #pragma unroll
        for (int h = 0; h < HCONST; ++h) oh[h] = b2s[h];
        for (int j = 0; j < RPEH; ++j){
            float hv = fmaxf(c0*w1s[j] + c1*w1s[RPEH + j] + b1s[j], 0.f);
            #pragma unroll
            for (int h = 0; h < HCONST; ++h) oh[h] += hv * w2s[j*HCONST + h];
        }
        #pragma unroll
        for (int h = 0; h < HCONST; ++h) sc[h][k] = oh[h];
    }
    __syncthreads();
    // phase 2: cosine dots (wave per head)
    int h = tid >> 5, d = tid & 31;
    int bbase = (row >> 10) << 10;   // b*T
    float qv = qrow[h*HD + d];
    for (int k = 0; k < TCONST; ++k){
        float p = qv * Kn[(size_t)(bbase + k)*MD + h*HD + d];
        #pragma unroll
        for (int off = 16; off; off >>= 1) p += __shfl_xor(p, off, 32);
        if (d == 0) sc[h][k] = p * scl[h] + sc[h][k];
    }
    __syncthreads();
    // phase 3: softmax per head
    float mx = -1e30f;
    for (int k = d; k < TCONST; k += 32) mx = fmaxf(mx, sc[h][k]);
    #pragma unroll
    for (int off = 16; off; off >>= 1) mx = fmaxf(mx, __shfl_xor(mx, off, 32));
    float sm = 0.f;
    for (int k = d; k < TCONST; k += 32){ float e = __expf(sc[h][k] - mx); sc[h][k] = e; sm += e; }
    #pragma unroll
    for (int off = 16; off; off >>= 1) sm += __shfl_xor(sm, off, 32);
    float inv = 1.f / sm;
    for (int k = d; k < TCONST; k += 32) sc[h][k] *= inv;
    __syncthreads();
    // phase 4: att @ V
    float a = 0.f;
    for (int k = 0; k < TCONST; ++k) a += sc[h][k] * V[(size_t)(bbase + k)*MD + h*HD + d];
    Out[(size_t)row*MD + tid] = a;
}

// ---------------------------------------------------------------------------
// top-16 nearest neighbors (argmin selection x16) + gather rel-coords
// ---------------------------------------------------------------------------
__global__ __launch_bounds__(256)
void topk_kernel(const float* __restrict__ rel, int* __restrict__ idxOut,
                 float* __restrict__ rcnb){
    __shared__ float dist[TCONST];
    __shared__ float rv[256]; __shared__ int ri[256];
    int row = blockIdx.x, tid = threadIdx.x;
    const float* rc = rel + (size_t)row*TCONST*2;
    for (int k = tid; k < TCONST; k += 256){ float a = rc[2*k], b = rc[2*k+1]; dist[k] = a*a + b*b; }
    __syncthreads();
    for (int s = 0; s < NHD; ++s){
        float bv = 1e30f; int bi = TCONST;
        for (int k = tid; k < TCONST; k += 256){
            float v = dist[k];
            if (v < bv || (v == bv && k < bi)){ bv = v; bi = k; }
        }
        rv[tid] = bv; ri[tid] = bi; __syncthreads();
        for (int off = 128; off; off >>= 1){
            if (tid < off){
                float ov = rv[tid+off]; int oi = ri[tid+off];
                if (ov < rv[tid] || (ov == rv[tid] && oi < ri[tid])){ rv[tid] = ov; ri[tid] = oi; }
            }
            __syncthreads();
        }
        if (tid == 0){
            int w = ri[0];
            idxOut[row*NHD + s] = w;
            dist[w] = 1e30f;
            rcnb[(size_t)(row*NHD + s)*2]     = rc[2*w];
            rcnb[(size_t)(row*NHD + s)*2 + 1] = rc[2*w + 1];
        }
        __syncthreads();
    }
}

// ---------------------------------------------------------------------------
// gather x_nb (f32 + f16)
// ---------------------------------------------------------------------------
__global__ void gather_kernel(const float* __restrict__ X3, const int* __restrict__ idx,
                              float* __restrict__ Xf, _Float16* __restrict__ Xfh){
    size_t i = (size_t)blockIdx.x*256 + threadIdx.x;
    if (i >= (size_t)LROWS*MD) return;
    int col = (int)(i % MD);
    int r   = (int)(i / MD);       // bt*16 + j
    int bt  = r >> 4;
    int b   = bt >> 10;
    int src = idx[r];
    float v = X3[((size_t)(b << 10) + src)*MD + col];
    Xf[i] = v; Xfh[i] = (_Float16)v;
}

// ---------------------------------------------------------------------------
// Local 16-token attention, fused RPE-bias MLP. One block (256 thr) per bt.
// ---------------------------------------------------------------------------
__global__ __launch_bounds__(256)
void lattn_kernel(const float* __restrict__ Qn, const float* __restrict__ Kn,
                  const float* __restrict__ V,  const float* __restrict__ rcnb,
                  const float* __restrict__ w1, const float* __restrict__ b1,
                  const float* __restrict__ w2, const float* __restrict__ b2,
                  const float* __restrict__ ls, float* __restrict__ Out){
    __shared__ float q_s[NHD][MD], k_s[NHD][MD], v_s[NHD][MD];
    __shared__ float sc[HCONST][NHD][NHD];
    __shared__ float rcs[NHD][2];
    __shared__ float w1s[2*RPEH], b1s[RPEH], w2s[RPEH*HCONST], b2s[HCONST], scl[HCONST];
    int bt = blockIdx.x, tid = threadIdx.x;
    size_t base = (size_t)bt*NHD*MD;
    for (int i = tid; i < NHD*MD; i += 256){
        q_s[i/MD][i%MD] = Qn[base + i];
        k_s[i/MD][i%MD] = Kn[base + i];
        v_s[i/MD][i%MD] = V[base + i];
    }
    for (int i = tid; i < NHD*2; i += 256) rcs[i/2][i%2] = rcnb[(size_t)bt*NHD*2 + i];
    for (int i = tid; i < 2*RPEH; i += 256) w1s[i] = w1[i];
    for (int i = tid; i < RPEH; i += 256) b1s[i] = b1[i];
    for (int i = tid; i < RPEH*HCONST; i += 256) w2s[i] = w2[i];
    if (tid < HCONST){ b2s[tid] = b2[tid]; scl[tid] = fminf(__expf(ls[tid]), 100.f); }
    __syncthreads();
    // bias MLP + scores : thread = (qi,kj)
    int qi = tid >> 4, kj = tid & 15;
    float c0 = rcs[kj][0] - rcs[qi][0];
    float c1 = rcs[kj][1] - rcs[qi][1];
    float oh[HCONST];
    #pragma unroll
    for (int h = 0; h < HCONST; ++h) oh[h] = b2s[h];
    for (int j = 0; j < RPEH; ++j){
        float hv = fmaxf(c0*w1s[j] + c1*w1s[RPEH + j] + b1s[j], 0.f);
        #pragma unroll
        for (int h = 0; h < HCONST; ++h) oh[h] += hv * w2s[j*HCONST + h];
    }
    #pragma unroll
    for (int h = 0; h < HCONST; ++h){
        float dot = 0.f;
        #pragma unroll
        for (int d = 0; d < HD; ++d) dot += q_s[qi][h*HD + d] * k_s[kj][h*HD + d];
        sc[h][qi][kj] = dot * scl[h] + oh[h];
    }
    __syncthreads();
    // softmax rows (h,i)
    if (tid < HCONST*NHD){
        int h = tid / NHD, i = tid % NHD;
        float mx = -1e30f;
        #pragma unroll
        for (int j = 0; j < NHD; ++j) mx = fmaxf(mx, sc[h][i][j]);
        float sm = 0.f;
        #pragma unroll
        for (int j = 0; j < NHD; ++j){ float e = __expf(sc[h][i][j] - mx); sc[h][i][j] = e; sm += e; }
        float inv = 1.f / sm;
        #pragma unroll
        for (int j = 0; j < NHD; ++j) sc[h][i][j] *= inv;
    }
    __syncthreads();
    // att @ V
    for (int o = tid; o < NHD*MD; o += 256){
        int i = o / MD, col = o % MD, h = col >> 5;
        float a = 0.f;
        #pragma unroll
        for (int j = 0; j < NHD; ++j) a += sc[h][i][j] * v_s[j][col];
        Out[base + o] = a;
    }
}

// ---------------------------------------------------------------------------
// Fused local FFN: s[row] = lrelu( sum_n lrelu((Xf2 @ W1)[row,n]) * w2[n] )
// One wave per 16 rows; 10 resident A-fragments (pre-swizzled, 32B loads);
// B fragments software-pipelined one deep (load bnext while WMMA uses bcur).
// W1sw is padded by one tile so the last prefetch stays in bounds.
// ---------------------------------------------------------------------------
__global__ __launch_bounds__(32)
void local_ffn_kernel(const _Float16* __restrict__ Aswz, const _Float16* __restrict__ W1sw,
                      const float* __restrict__ w2, float* __restrict__ srow){
    int lane = threadIdx.x & 31;
    int r16 = lane & 15, hi = lane >> 4;
    int m0 = blockIdx.x * 16;
    const v16h* Af = (const v16h*)Aswz;
    const v16h* Wf = (const v16h*)W1sw;
    v16h afr[10];
    #pragma unroll
    for (int kt = 0; kt < 10; ++kt)
        afr[kt] = Af[((size_t)blockIdx.x*10 + kt)*32 + lane];
    float rs[8] = {0.f,0.f,0.f,0.f,0.f,0.f,0.f,0.f};
    v16h bcur = Wf[lane];                      // tile (nt=0,kt=0)
    for (int nt = 0; nt < FF/16; ++nt){
        v8f c = {};
        #pragma unroll
        for (int kt = 0; kt < 10; ++kt){
            v16h bnext = Wf[((size_t)(nt*10 + kt + 1))*32 + lane];  // padded buffer
            c = __builtin_amdgcn_wmma_f32_16x16x32_f16(false, afr[kt], false, bcur,
                                                       (short)0, c, false, false);
            bcur = bnext;
        }
        float w2c = w2[nt*16 + r16];
        #pragma unroll
        for (int r = 0; r < 8; ++r) rs[r] += lrelu(c[r]) * w2c;
    }
    #pragma unroll
    for (int r = 0; r < 8; ++r)
        #pragma unroll
        for (int off = 1; off < 16; off <<= 1) rs[r] += __shfl_xor(rs[r], off, 32);
    if (r16 == 0){
        #pragma unroll
        for (int r = 0; r < 8; ++r) srow[m0 + r + 8*hi] = lrelu(rs[r]);
    }
}

// ---------------------------------------------------------------------------
// Y[i] = X[i] + s[i/MD]
// ---------------------------------------------------------------------------
__global__ void addrow_kernel(const float* __restrict__ X, const float* __restrict__ s,
                              float* __restrict__ Y, size_t n){
    size_t i = (size_t)blockIdx.x*256 + threadIdx.x;
    if (i < n) Y[i] = X[i] + s[i / MD];
}

// ============================================================================
// Host orchestration
// ============================================================================
static inline size_t cdiv(size_t a, size_t b){ return (a + b - 1) / b; }

extern "C" void kernel_launch(void* const* d_in, const int* in_sizes, int n_in,
                              void* d_out, int out_size, void* d_ws, size_t ws_size,
                              hipStream_t stream) {
    (void)in_sizes; (void)n_in; (void)out_size; (void)ws_size;
    // ----- input mapping (setup_inputs dict traversal order) -----
    const float* x_in   = (const float*)d_in[0];
    const float* rel    = (const float*)d_in[1];
    const float* g_n1_g = (const float*)d_in[2];
    const float* g_n1_b = (const float*)d_in[3];
    const float* g_n2_g = (const float*)d_in[4];
    const float* g_n2_b = (const float*)d_in[5];
    const float* g_wq   = (const float*)d_in[6];
    const float* g_wk   = (const float*)d_in[7];
    const float* g_wv   = (const float*)d_in[8];
    const float* g_wo   = (const float*)d_in[9];
    const float* g_ls   = (const float*)d_in[10];
    const float* g_rw1  = (const float*)d_in[11];
    const float* g_rb1  = (const float*)d_in[12];
    const float* g_rw2  = (const float*)d_in[13];
    const float* g_rb2  = (const float*)d_in[14];
    const float* g_w1   = (const float*)d_in[15];
    const float* g_w2   = (const float*)d_in[16];
    const float* l_n1_g = (const float*)d_in[17];
    const float* l_n1_b = (const float*)d_in[18];
    const float* l_n2_g = (const float*)d_in[19];
    const float* l_n2_b = (const float*)d_in[20];
    const float* l_n3_g = (const float*)d_in[21];
    const float* l_n3_b = (const float*)d_in[22];
    const float* l_wq   = (const float*)d_in[23];
    const float* l_wk   = (const float*)d_in[24];
    const float* l_wv   = (const float*)d_in[25];
    const float* l_wo   = (const float*)d_in[26];
    const float* l_ls   = (const float*)d_in[27];
    const float* l_rw1  = (const float*)d_in[28];
    const float* l_rb1  = (const float*)d_in[29];
    const float* l_rw2  = (const float*)d_in[30];
    const float* l_rb2  = (const float*)d_in[31];
    const float* l_w1   = (const float*)d_in[32];
    const float* l_w2   = (const float*)d_in[33];
    const float* u_w1   = (const float*)d_in[34];
    const float* u_w2   = (const float*)d_in[35];
    float* out = (float*)d_out;

    // ----- workspace bump allocator -----
    char* ws = (char*)d_ws; size_t off = 0;
    auto alloc = [&](size_t bytes)->void*{
        off = (off + 255) & ~(size_t)255;
        void* p = ws + off; off += bytes; return p;
    };
    // f16 weights
    _Float16* g_wq16 = (_Float16*)alloc(MD*MD*2);
    _Float16* g_wk16 = (_Float16*)alloc(MD*MD*2);
    _Float16* g_wv16 = (_Float16*)alloc(MD*MD*2);
    _Float16* g_wo16 = (_Float16*)alloc(MD*MD*2);
    _Float16* g_w116 = (_Float16*)alloc((size_t)MD*FF*2);
    _Float16* g_w216 = (_Float16*)alloc((size_t)FF*MD*2);
    _Float16* l_wq16 = (_Float16*)alloc(MD*MD*2);
    _Float16* l_wk16 = (_Float16*)alloc(MD*MD*2);
    _Float16* l_wv16 = (_Float16*)alloc(MD*MD*2);
    _Float16* l_wo16 = (_Float16*)alloc(MD*MD*2);
    _Float16* l_w1sw = (_Float16*)alloc(((size_t)MD*FF + 512)*2);  // +1 tile pad
    _Float16* u_w116 = (_Float16*)alloc((size_t)UD*UD*2);
    _Float16* u_w216 = (_Float16*)alloc((size_t)UD*MD*2);
    // global-stage activations
    float*    xn     = (float*)   alloc((size_t)GROWS*MD*4);
    _Float16* xn16   = (_Float16*)alloc((size_t)GROWS*MD*2);
    float*    qg     = (float*)   alloc((size_t)GROWS*MD*4);
    float*    kg     = (float*)   alloc((size_t)GROWS*MD*4);
    float*    vg     = (float*)   alloc((size_t)GROWS*MD*4);
    float*    attg   = (float*)   alloc((size_t)GROWS*MD*4);
    _Float16* attg16 = (_Float16*)alloc((size_t)GROWS*MD*2);
    float*    x1     = (float*)   alloc((size_t)GROWS*MD*4);
    float*    xn2    = (float*)   alloc((size_t)GROWS*MD*4);
    _Float16* xn2_16 = (_Float16*)alloc((size_t)GROWS*MD*2);
    _Float16* gh16   = (_Float16*)alloc((size_t)GROWS*FF*2);
    float*    x2     = (float*)   alloc((size_t)GROWS*MD*4);
    float*    x3     = (float*)   alloc((size_t)GROWS*MD*4);
    int*      idx    = (int*)     alloc((size_t)GROWS*NHD*4);
    float*    rcnb   = (float*)   alloc((size_t)GROWS*NHD*2*4);
    float*    srow   = (float*)   alloc((size_t)LROWS*4);
    // big reusable pools (5 x f32, 3 x f16 of LROWS*MD)
    const size_t BIGF = (size_t)LROWS*MD*4, BIGH = (size_t)LROWS*MD*2;
    float* BF0 = (float*)alloc(BIGF); float* BF1 = (float*)alloc(BIGF);
    float* BF2 = (float*)alloc(BIGF); float* BF3 = (float*)alloc(BIGF);
    float* BF4 = (float*)alloc(BIGF);
    _Float16* BH0 = (_Float16*)alloc(BIGH); _Float16* BH1 = (_Float16*)alloc(BIGH);
    _Float16* BH2 = (_Float16*)alloc(BIGH);
    float*    xf = BF0, *lq = BF1, *lk = BF2, *lv = BF3, *latt = BF4;
    float*    xf1 = BF1, *xf2 = BF2, *xf3 = BF3;
    _Float16* xf16 = BH0, *latt16 = BH1, *xf2A = BH0, *xu16 = BH1, *uh16 = BH2;

    // ----- weight casts / repacks -----
    auto cast = [&](const float* src, _Float16* dst, size_t n){
        cast_f16_kernel<<<dim3((unsigned)cdiv(n,256)), dim3(256), 0, stream>>>(src, dst, n);
    };
    cast(g_wq, g_wq16, MD*MD);  cast(g_wk, g_wk16, MD*MD);
    cast(g_wv, g_wv16, MD*MD);  cast(g_wo, g_wo16, MD*MD);
    cast(g_w1, g_w116, (size_t)MD*FF); cast(g_w2, g_w216, (size_t)FF*MD);
    cast(l_wq, l_wq16, MD*MD);  cast(l_wk, l_wk16, MD*MD);
    cast(l_wv, l_wv16, MD*MD);  cast(l_wo, l_wo16, MD*MD);
    repack_w1_kernel<<<dim3((unsigned)cdiv((size_t)MD*FF,256)),256,0,stream>>>(l_w1, l_w1sw);
    cast(u_w1, u_w116, (size_t)UD*UD); cast(u_w2, u_w216, (size_t)UD*MD);

    auto gemm = [&](bool lr, const _Float16* A, const _Float16* B, float* Cf,
                    _Float16* Ch, const float* R, int M, int N, int K){
        dim3 grid((unsigned)(M/128), (unsigned)(N/64));
        if (Ch){ if (lr) gemm_wmma_kernel<true ,true ><<<grid,256,0,stream>>>(A,B,nullptr,Ch,R,M,N,K);
                 else    gemm_wmma_kernel<false,true ><<<grid,256,0,stream>>>(A,B,nullptr,Ch,R,M,N,K); }
        else   { if (lr) gemm_wmma_kernel<true ,false><<<grid,256,0,stream>>>(A,B,Cf,nullptr,R,M,N,K);
                 else    gemm_wmma_kernel<false,false><<<grid,256,0,stream>>>(A,B,Cf,nullptr,R,M,N,K); }
    };

    // ================= global block =================
    ln_kernel<<<GROWS,256,0,stream>>>(x_in, g_n1_g, g_n1_b, xn, xn16, MD);
    gemm(false, xn16, g_wq16, qg, nullptr, nullptr, GROWS, MD, MD);
    gemm(false, xn16, g_wk16, kg, nullptr, nullptr, GROWS, MD, MD);
    gemm(false, xn16, g_wv16, vg, nullptr, nullptr, GROWS, MD, MD);
    qknorm_kernel<<<GROWS,320,0,stream>>>(qg);
    qknorm_kernel<<<GROWS,320,0,stream>>>(kg);
    gattn_kernel<<<GROWS,320,0,stream>>>(qg, kg, vg, rel, g_rw1, g_rb1, g_rw2, g_rb2, g_ls, attg);
    cast(attg, attg16, (size_t)GROWS*MD);
    gemm(false, attg16, g_wo16, x1, nullptr, xn, GROWS, MD, MD);     // x1 = xn + o-proj
    ln_kernel<<<GROWS,256,0,stream>>>(x1, g_n2_g, g_n2_b, xn2, xn2_16, MD);
    gemm(true , xn2_16, g_w116, nullptr, gh16, nullptr, GROWS, FF, MD);   // lrelu -> f16
    gemm(true , gh16,   g_w216, x2, nullptr, xn2, GROWS, MD, FF);         // x2 = xn2 + lrelu(.)

    // ================= local block =================
    ln_kernel<<<GROWS,256,0,stream>>>(x2, l_n1_g, l_n1_b, x3, nullptr, MD);
    topk_kernel<<<GROWS,256,0,stream>>>(rel, idx, rcnb);
    gather_kernel<<<dim3((unsigned)cdiv((size_t)LROWS*MD,256)),256,0,stream>>>(x3, idx, xf, xf16);
    gemm(false, xf16, l_wq16, lq, nullptr, nullptr, LROWS, MD, MD);
    gemm(false, xf16, l_wk16, lk, nullptr, nullptr, LROWS, MD, MD);
    gemm(false, xf16, l_wv16, lv, nullptr, nullptr, LROWS, MD, MD);
    qknorm_kernel<<<LROWS,320,0,stream>>>(lq);
    qknorm_kernel<<<LROWS,320,0,stream>>>(lk);
    lattn_kernel<<<GROWS,256,0,stream>>>(lq, lk, lv, rcnb, l_rw1, l_rb1, l_rw2, l_rb2, l_ls, latt);
    cast(latt, latt16, (size_t)LROWS*MD);
    gemm(false, latt16, l_wo16, xf1, nullptr, xf, LROWS, MD, MD);    // xf1 = xf + o-proj
    ln_kernel<<<LROWS,256,0,stream>>>(xf1, l_n2_g, l_n2_b, xf2, nullptr, MD);
    repack_a_kernel<<<dim3((unsigned)cdiv((size_t)LROWS*MD,256)),256,0,stream>>>(xf2, xf2A,
                                                                                 (size_t)LROWS*MD);
    local_ffn_kernel<<<LROWS/16,32,0,stream>>>(xf2A, l_w1sw, l_w2, srow);
    addrow_kernel<<<dim3((unsigned)cdiv((size_t)LROWS*MD,256)),256,0,stream>>>(xf2, srow, xf3,
                                                                               (size_t)LROWS*MD);
    // xu = reshape(xf3, [4096,5120]); LN3 -> f16
    ln_kernel<<<GROWS,256,0,stream>>>(xf3, l_n3_g, l_n3_b, nullptr, xu16, UD);
    gemm(true , xu16, u_w116, nullptr, uh16, nullptr, GROWS, UD, UD);     // lrelu -> f16
    gemm(true , uh16, u_w216, out, nullptr, nullptr, GROWS, MD, UD);      // final lrelu -> d_out
}
